// InterpolationPredictionModel_82781199663305
// MI455X (gfx1250) — matrix operand order repaired and checked by reading, not compiled, not run on hardware
//
#include <hip/hip_runtime.h>
#include <hip/hip_bf16.h>
#include <math.h>

// ---------------- problem constants (match reference) ----------------
#define NB   8      // B
#define NT   384    // T
#define ND   32     // D
#define NG   193    // G
#define NU   256    // U
#define NDEMO 16
#define NO   2
#define K_IH 96     // 3*D
#define N_OUT 768   // 3*U
#define MTOT (NB*NG)        // 1544 rows of the big GEMM
#define MTILES 97           // ceil(1544/16)
#define NTILES 48           // 768/16
#define KT_IH 3             // 96/32
#define KT_HH 8             // 256/32
#define LOG_EPS  (-20.7232658f)   // log(1e-9)
#define LOG_ONE  (1.0e-9f)        // log(1 + 1e-9)

typedef __attribute__((ext_vector_type(16))) _Float16 v16h;
typedef __attribute__((ext_vector_type(8)))  _Float16 v8h;
typedef __attribute__((ext_vector_type(8)))  float    v8f;
typedef __attribute__((ext_vector_type(4)))  float    v4f;

// ---------------- workspace layout (float units) ----------------
#define OFF_VALS  0                               // [B,D,T]
#define OFF_MASK  (OFF_VALS + NB*ND*NT)           // [B,D,T]
#define OFF_SIC   (OFF_MASK + NB*ND*NT)           // [B,3D,G]
#define OFF_RNN   (OFF_SIC + NB*3*ND*NG)          // [B*G,96]
#define OFF_XALL  (OFF_RNN + MTOT*K_IH)           // [B*G,768]
#define OFF_H     (OFF_XALL + (size_t)MTOT*N_OUT) // [B,U]
#define OFF_MEANG (OFF_H + NB*NU)                 // [B,D]
#define OFF_MEANT (OFF_MEANG + NB*ND)             // [B,D]
#define OFF_YC    (OFF_MEANT + NB*ND)             // [B,D,T]
#define OFF_WC    (OFF_YC + NB*ND*NT)             // [B,D,T]
#define OFF_CTX   (OFF_WC + NB*ND*NT)             // [B,D,T]
#define OFF_SE    (OFF_CTX + NB*ND*NT)            // [8] se sums, then [8] counts
#define OFF_PACK  (OFF_SE + 32)                   // f16 packed weights (8-float aligned)
#define PACK_IH_HALFS (NTILES*KT_IH*32*16)        // 73728
#define PACK_HH_HALFS (NTILES*KT_HH*32*16)        // 196608

// ===================== K0: preprocess / transpose / fixup ==============
__global__ void k_preprocess(const float* __restrict__ x,
                             const unsigned char* __restrict__ smask,
                             float* __restrict__ vals, float* __restrict__ maskf,
                             float* __restrict__ seBuf) {
  int b = blockIdx.x, t = threadIdx.x;         // 384 threads
  __shared__ int present[33];
  if (t < 33) present[t] = 0;
  if (b == 0 && t < 16) seBuf[t] = 0.0f;       // zero se sums + counts
  __syncthreads();
  int nun = 0;
  for (int d = 0; d < ND; ++d)
    nun += smask[((size_t)b*NT + t)*ND + d] ? 0 : 1;
  if (nun > 32) nun = 32;
  present[nun] = 1;                            // benign race, same value
  __syncthreads();
  for (int d = 0; d < ND; ++d) {
    float v  = x[((size_t)b*NT + t)*ND + d];
    float mk = smask[((size_t)b*NT + t)*ND + d] ? 1.f : 0.f;
    if (d == 0) {
      int aff = (t < 33) ? present[t] : 0;
      if (aff) { v = 0.f; mk = 1.f; }
    }
    vals[((size_t)b*ND + d)*NT + t]  = v;
    maskf[((size_t)b*ND + d)*NT + t] = mk;
  }
}

// ===================== K1: pack weights to WMMA B-fragment layout ======
// B fragment (16-bit, K x 16 tile): element j of lane half h holds
// K=kt*32+8h+j (j<8) or K=kt*32+16+8h+(j-8) (j>=8); N = nt*16 + lane%16.
__global__ void k_pack_b(const float* __restrict__ src, _Float16* __restrict__ dst,
                         int nKT, int K, int total) {
  int idx = blockIdx.x*blockDim.x + threadIdx.x;
  if (idx >= total) return;
  int j    = idx & 15;
  int lane = (idx >> 4) & 31;
  int kt   = (idx >> 9) % nKT;
  int nt   = idx / (nKT*512);
  int n = nt*16 + (lane & 15);
  int k = kt*32 + (lane >> 4)*8 + (j & 7) + (j >> 3)*16;
  dst[idx] = (_Float16)src[(size_t)n*K + k];
}

// ===================== K2: SCI onto the regular grid ===================
__global__ void k_sci_grid(const float* __restrict__ vals, const float* __restrict__ maskf,
                           const float* __restrict__ timep, const float* __restrict__ gridp,
                           const float* __restrict__ kern, float* __restrict__ sic) {
  int bd = blockIdx.x, b = bd >> 5, d = bd & 31;
  int tid = threadIdx.x;
  __shared__ float tS[NT], vS[NT], lS[NT];
  for (int t = tid; t < NT; t += 256) {
    tS[t] = timep[b*NT + t];
    vS[t] = vals[((size_t)b*ND + d)*NT + t];
    lS[t] = (maskf[((size_t)b*ND + d)*NT + t] > 0.5f) ? LOG_ONE : LOG_EPS;
  }
  __syncthreads();
  float kv = kern[d];
  float alpha = (kv > 20.f) ? kv : log1pf(__expf(kv));
  if (tid < NG) {
    float gv = gridp[tid];
    float outy[2], outw[2];
    for (int p = 0; p < 2; ++p) {
      float a = (p == 0) ? alpha : 10.f*alpha;
      float mx = -3.4e38f;
      for (int t = 0; t < NT; ++t) {
        float dt = tS[t] - gv;
        mx = fmaxf(mx, -a*dt*dt + lS[t]);
      }
      float s = 0.f, y = 0.f;
      for (int t = 0; t < NT; ++t) {
        float dt = tS[t] - gv;
        float e = __expf(-a*dt*dt + lS[t] - mx);
        s += e; y += e*vS[t];
      }
      outw[p] = mx + __logf(s);
      outy[p] = y / s;
    }
    size_t base = (size_t)b*3*ND*NG;
    sic[base + (size_t)d*NG + tid]        = outy[0];
    sic[base + (size_t)(ND+d)*NG + tid]   = outw[0];
    sic[base + (size_t)(2*ND+d)*NG + tid] = outy[1];
  }
}

// ===================== K3: per-channel mean over positions =============
__global__ void k_chan_mean(const float* __restrict__ src, float* __restrict__ dst,
                            int len, int bstride, int dstride) {
  int i = threadIdx.x;            // 256 = B*D
  int b = i >> 5, d = i & 31;
  const float* p = src + (size_t)b*bstride + (size_t)d*dstride;
  float s = 0.f;
  for (int t = 0; t < len; ++t) s += p[t];
  dst[i] = s / (float)len;
}

// ===================== K4: CCI on grid -> rnn_in =======================
__global__ void k_cci_grid(const float* __restrict__ sic, const float* __restrict__ meanG,
                           const float* __restrict__ cw, float* __restrict__ rnnin) {
  int tid = threadIdx.x, lane = tid & 31, wv = tid >> 5;
  int p = blockIdx.x*8 + wv;            // 0..1543 == b*G+g
  int b = p / NG, g = p % NG;
  size_t base = (size_t)b*3*ND*NG;
  float y   = sic[base + (size_t)lane*NG + g];
  float w   = sic[base + (size_t)(ND+lane)*NG + g];
  float y10 = sic[base + (size_t)(2*ND+lane)*NG + g];
  float m = w;
  for (int o = 16; o; o >>= 1) m = fmaxf(m, __shfl_xor(m, o, 32));
  float e = __expf(w - m), s = e;
  for (int o = 16; o; o >>= 1) s += __shfl_xor(s, o, 32);
  float soft = e / s;
  float mu = meanG[b*ND + lane];
  float sval = soft * (y - mu);
  float acc = 0.f;
  for (int d2 = 0; d2 < 32; ++d2)
    acc += __shfl(sval, d2, 32) * cw[d2*ND + lane];
  float rep1 = acc + mu;
  size_t rb = (size_t)p * K_IH;
  rnnin[rb + lane]      = rep1;
  rnnin[rb + 32 + lane] = __expf(w);
  rnnin[rb + 64 + lane] = y10 - rep1;
}

// ===================== K5: xall = rnn_in @ w_ih^T + b_ih (WMMA) ========
// A fragment (16-bit layout): lane's 16 halves are two contiguous runs of 8:
// k = kt*32 + khalf*8 + 0..7 and k = kt*32 + 16 + khalf*8 + 0..7.
// Out-of-range rows are CLAMPED (not predicated): a garbage A row only
// affects the matching C row; only the LAST M-tile has partial rows, and
// there the valid rows are exactly the khalf==0 half-wave.
__global__ void k_gemm_xall(const float* __restrict__ rnnin,
                            const _Float16* __restrict__ packIH,
                            const float* __restrict__ b_ih, float* __restrict__ xall) {
  int tid = threadIdx.x, lane = tid & 31, wv = tid >> 5;
  int gw = blockIdx.x*8 + wv;           // wave -> output tile
  if (gw >= MTILES*NTILES) return;
  int mt = gw / NTILES, nt = gw % NTILES;
  int lane16 = lane & 15, khalf = lane >> 4;
  const v16h* Bp = (const v16h*)packIH;
  v8f acc = {};
  int m = mt*16 + lane16;
  int mc = (m < MTOT) ? m : (MTOT - 1);           // clamp, no divergence
  const v4f* rp = (const v4f*)(rnnin + (size_t)mc*K_IH);  // 16B-aligned rows
#pragma unroll
  for (int kt = 0; kt < KT_IH; ++kt) {
    v4f a0 = rp[kt*8 + khalf*2];
    v4f a1 = rp[kt*8 + khalf*2 + 1];
    v4f a2 = rp[kt*8 + khalf*2 + 4];
    v4f a3 = rp[kt*8 + khalf*2 + 5];
    v16h a;
#pragma unroll
    for (int j = 0; j < 4; ++j) {
      a[j]      = (_Float16)a0[j];
      a[4 + j]  = (_Float16)a1[j];
      a[8 + j]  = (_Float16)a2[j];
      a[12 + j] = (_Float16)a3[j];
    }
    v16h bfr = Bp[(nt*KT_IH + kt)*32 + lane];
    acc = __builtin_amdgcn_wmma_f32_16x16x32_f16(false, a, false, bfr,
                                                 (short)0, acc, false, false);
  }
  int n = nt*16 + lane16;
  float bias = b_ih[n];                           // row-invariant: load once
  float* outp = xall + (size_t)(mt*16 + khalf*8)*N_OUT + n;
  if (mt < MTILES - 1) {                          // full tile: no bound checks
#pragma unroll
    for (int r = 0; r < 8; ++r) outp[(size_t)r*N_OUT] = acc[r] + bias;
  } else if (khalf == 0) {                        // last tile: rows 1536..1543
#pragma unroll
    for (int r = 0; r < 8; ++r) outp[(size_t)r*N_OUT] = acc[r] + bias;
  }
}

// ===================== K6: demographic encoder -> h0 ===================
__global__ void k_demo_h0(const float* __restrict__ stat,
                          const float* __restrict__ w1, const float* __restrict__ b1,
                          const float* __restrict__ w2, const float* __restrict__ b2,
                          float* __restrict__ hbuf) {
  int b = blockIdx.x, u = threadIdx.x;
  __shared__ float t1[NU];
  float s = b1[u];
  for (int k = 0; k < NDEMO; ++k) s += stat[b*NDEMO + k]*w1[k*NU + u];
  t1[u] = fmaxf(s, 0.f);
  __syncthreads();
  float h = b2[u];
  for (int k = 0; k < NU; ++k) h += t1[k]*w2[k*NU + u];
  hbuf[b*NU + u] = h;
}

// ===================== K7: GRU scan (1 workgroup, WMMA per step) =======
// f32 master state hS drives the gate math; a packed f16 mirror hH feeds the
// WMMA A fragments as two aligned 16B LDS loads per k-tile. Rows 8..15 of
// the 16-row A tile are clamped duplicates; only khalf==0 stores C rows.
// With blockDim=256 the gate mapping collapses to (b=i, u=tid), so the
// b_hh biases are hoisted out of the 193-step scan entirely.
__global__ void k_gru(const float* __restrict__ xall,
                      const _Float16* __restrict__ packHH,
                      const float* __restrict__ b_hh, float* __restrict__ hbuf) {
  __shared__ float    hS[NB*NU];        // 8x256 f32 hidden state
  __shared__ _Float16 hH[NB*NU];        // f16 mirror, row-major [b][u]
  __shared__ float    ghS[NB*N_OUT];    // 8x768 recurrent pre-activations
  int tid = threadIdx.x;
  for (int i = tid; i < NB*NU; i += 256) {
    float h0 = hbuf[i];
    hS[i] = h0;
    hH[i] = (_Float16)h0;
  }
  __syncthreads();
  int lane = tid & 31, wave = tid >> 5;
  int lane16 = lane & 15, khalf = lane >> 4;
  int rowA = lane16 & 7;                // clamp padded rows 8..15 -> 0..7
  const v16h* Bp = (const v16h*)packHH;
  const v8h*  hp = (const v8h*)hH;      // 16B vectors of 8 halves
  float bh_r = b_hh[tid], bh_z = b_hh[NU + tid], bh_n = b_hh[2*NU + tid];

  for (int g = 0; g < NG; ++g) {
    // --- A fragments from the f16 mirror: 2 x ds_load_b128 per k-tile ---
    v16h afrag[KT_HH];
#pragma unroll
    for (int kt = 0; kt < KT_HH; ++kt) {
      int vbase = rowA*32 + kt*4 + khalf;      // v8h units: (row*256+kt*32+khalf*8)/8
      v8h lo = hp[vbase];
      v8h hi = hp[vbase + 2];                  // +16 halves
      afrag[kt] = __builtin_shufflevector(lo, hi, 0,1,2,3,4,5,6,7,
                                                  8,9,10,11,12,13,14,15);
    }
    // --- each wave: 6 N-tiles of gh = h @ w_hh^T ---
    for (int s = 0; s < 6; ++s) {
      int nt = wave*6 + s;
      v8f acc = {};
#pragma unroll
      for (int kt = 0; kt < KT_HH; ++kt) {
        v16h bfr = Bp[(nt*KT_HH + kt)*32 + lane];
        acc = __builtin_amdgcn_wmma_f32_16x16x32_f16(false, afrag[kt], false, bfr,
                                                     (short)0, acc, false, false);
      }
      if (khalf == 0) {                        // C rows 0..7 live in this half
#pragma unroll
        for (int r = 0; r < 8; ++r) ghS[r*N_OUT + nt*16 + lane16] = acc[r];
      }
    }
    __syncthreads();
    // --- gates: b = i, u = tid ---
    float hnew[NB];
#pragma unroll
    for (int i = 0; i < NB; ++i) {
      const float* xr = xall + ((size_t)(i*NG + g))*N_OUT + tid;
      if (g + 1 < NG) __builtin_prefetch(xr + N_OUT, 0, 1);
      float xrv = xr[0], xzv = xr[NU], xnv = xr[2*NU];
      float hrv = ghS[i*N_OUT + tid]        + bh_r;
      float hzv = ghS[i*N_OUT + NU + tid]   + bh_z;
      float hnv = ghS[i*N_OUT + 2*NU + tid] + bh_n;
      float rg = 1.f / (1.f + __expf(-(xrv + hrv)));
      float zg = 1.f / (1.f + __expf(-(xzv + hzv)));
      float ng = tanhf(xnv + rg*hnv);
      hnew[i] = (1.f - zg)*ng + zg*hS[i*NU + tid];
    }
    __syncthreads();
#pragma unroll
    for (int i = 0; i < NB; ++i) {
      int off = i*NU + tid;
      hS[off] = hnew[i];
      hH[off] = (_Float16)hnew[i];
    }
    __syncthreads();
  }
  for (int i = tid; i < NB*NU; i += 256) hbuf[i] = hS[i];
}

// ===================== K8: output projection ===========================
__global__ void k_out_proj(const float* __restrict__ hbuf, const float* __restrict__ ow,
                           const float* __restrict__ ob, float* __restrict__ out) {
  int t = threadIdx.x;
  if (t < NB*NO) {
    int b = t / NO, o = t % NO;
    float s = ob[o];
    for (int u = 0; u < NU; ++u) s += hbuf[b*NU + u]*ow[u*NO + o];
    out[t] = s;
  }
}

// ===================== K9: SCI reconstruction (ref points = times) =====
__global__ void k_sci_recon(const float* __restrict__ vals, const float* __restrict__ maskf,
                            const unsigned char* __restrict__ rmask,
                            const float* __restrict__ timep, const float* __restrict__ kern,
                            float* __restrict__ yc, float* __restrict__ wc,
                            float* __restrict__ ctxf) {
  int bd = blockIdx.x, b = bd >> 5, d = bd & 31;
  int tid = threadIdx.x;
  __shared__ float tS[NT], vS[NT], lS[NT], cS[NT];
  __shared__ int anyF;
  if (tid == 0) anyF = 0;
  __syncthreads();
  for (int t = tid; t < NT; t += 256) {
    tS[t] = timep[b*NT + t];
    vS[t] = vals[((size_t)b*ND + d)*NT + t];
    float mk = maskf[((size_t)b*ND + d)*NT + t];
    int rm = rmask[((size_t)b*NT + t)*ND + d] ? 1 : 0;
    int c = (mk > 0.5f) && rm;
    cS[t] = c ? 1.f : 0.f;
    if (c) anyF = 1;
  }
  __syncthreads();
  int any = anyF;
  for (int t = tid; t < NT; t += 256) {
    float c = any ? cS[t] : 1.f;              // fallback: all-context if empty
    cS[t] = c;
    lS[t] = (c > 0.5f) ? LOG_ONE : LOG_EPS;
    ctxf[((size_t)b*ND + d)*NT + t] = c;
  }
  __syncthreads();
  float kv = kern[d];
  float alpha = (kv > 20.f) ? kv : log1pf(__expf(kv));
  for (int to = tid; to < NT; to += 256) {
    float gv = tS[to];
    float mx = -3.4e38f;
    for (int t = 0; t < NT; ++t) {
      float dt = tS[t] - gv;
      mx = fmaxf(mx, -alpha*dt*dt + lS[t]);
    }
    float s = 0.f, y = 0.f;
    for (int t = 0; t < NT; ++t) {
      float dt = tS[t] - gv;
      float e = __expf(-alpha*dt*dt + lS[t] - mx);
      s += e; y += e*vS[t];
    }
    yc[((size_t)b*ND + d)*NT + to] = y / s;
    wc[((size_t)b*ND + d)*NT + to] = mx + __logf(s);
  }
}

// ===================== K10: CCI reconstruction + loss accumulation =====
__global__ void k_cci_recon_loss(const float* __restrict__ yc, const float* __restrict__ wc,
                                 const float* __restrict__ meanT, const float* __restrict__ cw,
                                 const float* __restrict__ vals, const float* __restrict__ maskf,
                                 const float* __restrict__ ctxf, float* __restrict__ seBuf) {
  int tid = threadIdx.x, lane = tid & 31, wv = tid >> 5;
  int p = blockIdx.x*8 + wv;            // 0..3071 == b*T+to
  int b = p / NT, to = p % NT;
  size_t rb = ((size_t)b*ND + lane)*NT + to;
  float y = yc[rb], w = wc[rb];
  float m = w;
  for (int o = 16; o; o >>= 1) m = fmaxf(m, __shfl_xor(m, o, 32));
  float e = __expf(w - m), s = e;
  for (int o = 16; o; o >>= 1) s += __shfl_xor(s, o, 32);
  float soft = e / s;
  float mu = meanT[b*ND + lane];
  float sval = soft * (y - mu);
  float acc = 0.f;
  for (int d2 = 0; d2 < 32; ++d2)
    acc += __shfl(sval, d2, 32) * cw[d2*ND + lane];
  float crc = acc + mu;                 // crc_r[b, lane, to]
  float tgt = (maskf[rb] > 0.5f && ctxf[rb] < 0.5f) ? 1.f : 0.f;
  float diff = vals[rb] - crc;
  float se = tgt*diff*diff;
  for (int o = 16; o; o >>= 1) { se += __shfl_xor(se, o, 32); tgt += __shfl_xor(tgt, o, 32); }
  if (lane == 0) { atomicAdd(&seBuf[b], se); atomicAdd(&seBuf[8 + b], tgt); }
}

// ===================== K11: finalize loss ==============================
__global__ void k_finalize(const float* __restrict__ seBuf, float* __restrict__ out) {
  if (threadIdx.x == 0) {
    float acc = 0.f;
    for (int b = 0; b < NB; ++b) acc += seBuf[b] / (seBuf[8 + b] + 1e-9f);
    out[NB*NO] = acc / (float)NB;
  }
}

// ===================== launch ==========================================
extern "C" void kernel_launch(void* const* d_in, const int* in_sizes, int n_in,
                              void* d_out, int out_size, void* d_ws, size_t ws_size,
                              hipStream_t stream) {
  const float* x        = (const float*)d_in[0];
  const float* stat     = (const float*)d_in[1];
  const float* timep    = (const float*)d_in[2];
  const float* gridp    = (const float*)d_in[3];
  const float* kern     = (const float*)d_in[4];
  const float* cross_w  = (const float*)d_in[5];
  const float* demo_w1  = (const float*)d_in[6];
  const float* demo_b1  = (const float*)d_in[7];
  const float* demo_w2  = (const float*)d_in[8];
  const float* demo_b2  = (const float*)d_in[9];
  const float* w_ih     = (const float*)d_in[10];
  const float* w_hh     = (const float*)d_in[11];
  const float* b_ih     = (const float*)d_in[12];
  const float* b_hh     = (const float*)d_in[13];
  const float* out_w    = (const float*)d_in[14];
  const float* out_b    = (const float*)d_in[15];
  const unsigned char* smask = (const unsigned char*)d_in[16];  // bool arrays: 1 byte/elem
  const unsigned char* rmask = (const unsigned char*)d_in[17];
  float* out = (float*)d_out;

  float* ws      = (float*)d_ws;
  float* vals    = ws + OFF_VALS;
  float* maskf   = ws + OFF_MASK;
  float* sic     = ws + OFF_SIC;
  float* rnnin   = ws + OFF_RNN;
  float* xall    = ws + OFF_XALL;
  float* hbuf    = ws + OFF_H;
  float* meanG   = ws + OFF_MEANG;
  float* meanT   = ws + OFF_MEANT;
  float* yc      = ws + OFF_YC;
  float* wc      = ws + OFF_WC;
  float* ctxf    = ws + OFF_CTX;
  float* seBuf   = ws + OFF_SE;
  _Float16* packIH = (_Float16*)(ws + OFF_PACK);
  _Float16* packHH = packIH + PACK_IH_HALFS;

  k_preprocess<<<NB, NT, 0, stream>>>(x, smask, vals, maskf, seBuf);
  k_pack_b<<<(PACK_IH_HALFS + 255)/256, 256, 0, stream>>>(w_ih, packIH, KT_IH, K_IH, PACK_IH_HALFS);
  k_pack_b<<<(PACK_HH_HALFS + 255)/256, 256, 0, stream>>>(w_hh, packHH, KT_HH, NU, PACK_HH_HALFS);
  k_sci_grid<<<NB*ND, 256, 0, stream>>>(vals, maskf, timep, gridp, kern, sic);
  k_chan_mean<<<1, 256, 0, stream>>>(sic, meanG, NG, 3*ND*NG, NG);
  k_cci_grid<<<NG, 256, 0, stream>>>(sic, meanG, cross_w, rnnin);
  k_gemm_xall<<<(MTILES*NTILES + 7)/8, 256, 0, stream>>>(rnnin, packIH, b_ih, xall);
  k_demo_h0<<<NB, NU, 0, stream>>>(stat, demo_w1, demo_b1, demo_w2, demo_b2, hbuf);
  k_gru<<<1, 256, 0, stream>>>(xall, packHH, b_hh, hbuf);
  k_out_proj<<<1, 32, 0, stream>>>(hbuf, out_w, out_b, out);
  k_sci_recon<<<NB*ND, 256, 0, stream>>>(vals, maskf, rmask, timep, kern, yc, wc, ctxf);
  k_chan_mean<<<1, 256, 0, stream>>>(yc, meanT, NT, ND*NT, NT);
  k_cci_recon_loss<<<NB*NT/8, 256, 0, stream>>>(yc, wc, meanT, cross_w, vals, maskf, ctxf, seBuf);
  k_finalize<<<1, 32, 0, stream>>>(seBuf, out);
}
